// TaylorAutoencoder_50525995270523
// MI455X (gfx1250) — matrix operand order repaired and checked by reading, not compiled, not run on hardware
//
#include <hip/hip_runtime.h>
#include <hip/hip_bf16.h>

typedef __attribute__((ext_vector_type(16))) _Float16 v16h;
typedef __attribute__((ext_vector_type(8)))  _Float16 v8h;
typedef __attribute__((ext_vector_type(8)))  float    v8f;
typedef __attribute__((ext_vector_type(4)))  int      v4i;

#define NROWS 8192
#define DIM   512
#define LDIM  32

#define NSPLIT 4
#define NT_PER ((NROWS / 16) / NSPLIT)   // 128 N-tiles per split
#define LROW   520                        // LDS row stride in halves (512 + 8 pad)

#if defined(__AMDGCN__) && __has_builtin(__builtin_amdgcn_global_load_async_to_lds_b128)
#define USE_ASYNC 1
typedef __attribute__((address_space(1))) v4i gv4i;   // global int4
typedef __attribute__((address_space(3))) v4i lv4i;   // LDS int4
#else
#define USE_ASYNC 0
#endif

// ---------------------------------------------------------------------------
// Kernel 1: row squared-norms (f32) + cast xs -> f16 workspace copy
// ---------------------------------------------------------------------------
__global__ __launch_bounds__(256) void prep_kernel(const float* __restrict__ xs,
                                                   _Float16* __restrict__ Xh,
                                                   float* __restrict__ sq) {
  const int row = blockIdx.x;
  const float* x = xs + (size_t)row * DIM;
  _Float16* xh = Xh + (size_t)row * DIM;
  float s = 0.0f;
  for (int i = threadIdx.x; i < DIM; i += 256) {
    float v = x[i];
    xh[i] = (_Float16)v;
    s = fmaf(v, v, s);
  }
#pragma unroll
  for (int off = 16; off >= 1; off >>= 1) s += __shfl_xor(s, off, 32);
  __shared__ float red[8];
  if ((threadIdx.x & 31) == 0) red[threadIdx.x >> 5] = s;
  __syncthreads();
  if (threadIdx.x == 0) {
    float t = 0.0f;
    for (int w = 0; w < 8; ++w) t += red[w];
    sq[row] = t;
  }
}

// ---------------------------------------------------------------------------
// Async staging of one 16x512 f16 B-tile into (padded) LDS.
// 256 threads x 4 chunks x 16B = 16 KB.
// ---------------------------------------------------------------------------
__device__ __forceinline__ void stage_tile(const _Float16* __restrict__ Xh,
                                           int n0, _Float16* lbuf, int tid) {
#pragma unroll
  for (int c = 0; c < 4; ++c) {
    const int chunk = tid * 4 + c;   // 0..1023
    const int rowi  = chunk >> 6;    // 0..15
    const int col8  = chunk & 63;    // 16-byte unit within row
    const _Float16* g = Xh + (size_t)(n0 + rowi) * DIM + col8 * 8;
    _Float16* l = lbuf + rowi * LROW + col8 * 8;
#if USE_ASYNC
    __builtin_amdgcn_global_load_async_to_lds_b128(
        (gv4i*)(unsigned long long)(const void*)g,
        (lv4i*)l, 0, 0);
#else
    *(v8h*)l = *(const v8h*)g;       // synchronous fallback
#endif
  }
}

__device__ __forceinline__ void wait_async_le4() {
#if USE_ASYNC
#if __has_builtin(__builtin_amdgcn_s_wait_asynccnt)
  __builtin_amdgcn_s_wait_asynccnt(4);
#else
  asm volatile("s_wait_asynccnt 0x4" ::: "memory");
#endif
#endif
}

__device__ __forceinline__ void wait_async_le0() {
#if USE_ASYNC
#if __has_builtin(__builtin_amdgcn_s_wait_asynccnt)
  __builtin_amdgcn_s_wait_asynccnt(0);
#else
  asm volatile("s_wait_asynccnt 0x0" ::: "memory");
#endif
#endif
}

// ---------------------------------------------------------------------------
// Kernel 2: fused Gram (WMMA f16->f32) + diagonal-masked argmin.
// Block = 8 waves, each wave owns a 16-row M-tile (block covers 128 rows).
// All waves sweep the SAME N-tiles; B tiles double-buffered in LDS via
// async global->LDS copies. Grid = NSPLIT x 64 blocks; partial argmin per
// split, combined by nn_reduce_kernel.
// ---------------------------------------------------------------------------
__global__ __launch_bounds__(256) void nn_kernel(const _Float16* __restrict__ Xh,
                                                 const float* __restrict__ sq,
                                                 float* __restrict__ pd,
                                                 int* __restrict__ pi) {
  __shared__ _Float16 lbuf[2][16 * LROW];   // ~32.5 KB

  const int split = blockIdx.x >> 6;        // 0..3
  const int mblk  = blockIdx.x & 63;        // 0..63
  const int tid   = threadIdx.x;
  const int lane  = tid & 31;
  const int wave  = tid >> 5;               // 0..7
  const int grp   = lane >> 4;              // K-group select (0/1)
  const int lr    = lane & 15;              // row (A) / column (B,C) in tile
  const int m0    = mblk * 128 + wave * 16;
  const int ntb   = split * NT_PER;

  // kick off the first B tile while we load A fragments
  stage_tile(Xh, ntb * 16, lbuf[0], tid);

  // --- A fragments for all 512 K (ISA 16-bit A 16x32 layout) ---------------
  v16h a[16];
  {
    const _Float16* arow = Xh + (size_t)(m0 + lr) * DIM;
#pragma unroll
    for (int ks = 0; ks < 16; ++ks) {
      const _Float16* p = arow + ks * 32 + grp * 8;
      v8h lo = *(const v8h*)(p);
      v8h hi = *(const v8h*)(p + 16);
      a[ks] = __builtin_shufflevector(lo, hi, 0, 1, 2, 3, 4, 5, 6, 7,
                                      8, 9, 10, 11, 12, 13, 14, 15);
    }
  }

  float sqm[8];
#pragma unroll
  for (int r = 0; r < 8; ++r) sqm[r] = sq[m0 + r + 8 * grp];

  float bestd[8];
  int   besti[8];
#pragma unroll
  for (int r = 0; r < 8; ++r) { bestd[r] = __builtin_inff(); besti[r] = 0x7fffffff; }

  for (int it = 0; it < NT_PER; ++it) {
    if (it + 1 < NT_PER) {
      stage_tile(Xh, (ntb + it + 1) * 16, lbuf[(it + 1) & 1], tid);
      wait_async_le4();   // in-order completion: previous tile's 4 copies done
    } else {
      wait_async_le0();
    }
    __syncthreads();      // tile (it) visible to all waves

    // B fragment from LDS: lane = column n (lr), K = 8*grp + 0..7 and 16 + 8*grp + 0..7
    const _Float16* lb = &lbuf[it & 1][lr * LROW + grp * 8];
    v8f acc = {};
#pragma unroll
    for (int ks = 0; ks < 16; ++ks) {
      v8h blo = *(const v8h*)(lb + ks * 32);
      v8h bhi = *(const v8h*)(lb + ks * 32 + 16);
      v16h b = __builtin_shufflevector(blo, bhi, 0, 1, 2, 3, 4, 5, 6, 7,
                                       8, 9, 10, 11, 12, 13, 14, 15);
      acc = __builtin_amdgcn_wmma_f32_16x16x32_f16(
          /*neg_a=*/false, a[ks], /*neg_b=*/false, b,
          /*c_mod=*/(short)0, acc, /*reuse_a=*/false, /*reuse_b=*/false);
    }

    const int   n0   = (ntb + it) * 16;
    const int   ncol = n0 + lr;
    const float sqn  = sq[ncol];
#pragma unroll
    for (int r = 0; r < 8; ++r) {
      const int mrow = m0 + r + 8 * grp;
      float d2 = fmaf(-2.0f, acc[r], sqm[r] + sqn);
      if (ncol == mrow) d2 = __builtin_inff();     // mask diagonal
      if (d2 < bestd[r] || (d2 == bestd[r] && ncol < besti[r])) {
        bestd[r] = d2;
        besti[r] = ncol;
      }
    }
    __syncthreads();      // done reading tile (it); buffer may be overwritten
  }

  // --- argmin across the 16 lanes of each group; rows are wave-private -----
#pragma unroll
  for (int r = 0; r < 8; ++r) {
    float d = bestd[r];
    int   i = besti[r];
#pragma unroll
    for (int off = 8; off >= 1; off >>= 1) {
      float od = __shfl_xor(d, off, 32);
      int   oi = __shfl_xor(i, off, 32);
      if (od < d || (od == d && oi < i)) { d = od; i = oi; }
    }
    if (lr == 0) {
      pd[(size_t)split * NROWS + m0 + r + 8 * grp] = d;
      pi[(size_t)split * NROWS + m0 + r + 8 * grp] = i;
    }
  }
}

// ---------------------------------------------------------------------------
// Kernel 2b: combine per-split argmin partials.
// ---------------------------------------------------------------------------
__global__ __launch_bounds__(256) void nn_reduce_kernel(const float* __restrict__ pd,
                                                        const int* __restrict__ pi,
                                                        int* __restrict__ closest) {
  const int m = blockIdx.x * 256 + threadIdx.x;
  float d = pd[m];
  int   i = pi[m];
  for (int s = 1; s < NSPLIT; ++s) {
    float od = pd[(size_t)s * NROWS + m];
    int   oi = pi[(size_t)s * NROWS + m];
    if (od < d || (od == d && oi < i)) { d = od; i = oi; }
  }
  closest[m] = i;
}

// ---------------------------------------------------------------------------
// Kernel 3: fused Taylor encoder (exact ReLU JVP) + decoder. One row / block.
// ---------------------------------------------------------------------------
__global__ __launch_bounds__(64) void mlp_kernel(
    const float* __restrict__ xs, const int* __restrict__ closest,
    const float* __restrict__ W1, const float* __restrict__ b1,
    const float* __restrict__ W2, const float* __restrict__ b2,
    const float* __restrict__ W3, const float* __restrict__ b3,
    const float* __restrict__ D1, const float* __restrict__ d1,
    const float* __restrict__ D2, const float* __restrict__ d2,
    const float* __restrict__ D3, const float* __restrict__ d3,
    float* __restrict__ xhat, float* __restrict__ zs_out) {
  __shared__ float xsh[DIM], x0sh[DIM];
  __shared__ float h1[64], t1g[64], h2[32], t2g[32], z[LDIM], g1[32], g2[64];

  const int row = blockIdx.x;
  const int t   = threadIdx.x;
  const int c   = closest[row];
  const float* x  = xs + (size_t)row * DIM;
  const float* x0 = xs + (size_t)c * DIM;
  for (int i = t; i < DIM; i += 64) { xsh[i] = x[i]; x0sh[i] = x0[i]; }
  __syncthreads();

  // layer 1: a1 = W1 x0 + b1 ; tangent t1 = W1 (x - x0), ReLU-gated at a1
  {
    const float* w = W1 + t * DIM;
    float u = 0.0f, v = 0.0f;
    for (int k = 0; k < DIM; ++k) {
      u = fmaf(w[k], xsh[k], u);
      v = fmaf(w[k], x0sh[k], v);
    }
    float aa = v + b1[t];
    h1[t]  = aa > 0.0f ? aa : 0.0f;
    t1g[t] = aa > 0.0f ? (u - v) : 0.0f;
  }
  __syncthreads();

  if (t < 32) {  // layer 2
    const float* w = W2 + t * 64;
    float u = 0.0f, v = 0.0f;
    for (int k = 0; k < 64; ++k) {
      v = fmaf(w[k], h1[k], v);
      u = fmaf(w[k], t1g[k], u);
    }
    float aa = v + b2[t];
    h2[t]  = aa > 0.0f ? aa : 0.0f;
    t2g[t] = aa > 0.0f ? u : 0.0f;
  }
  __syncthreads();

  if (t < 32) {  // layer 3 (linear): zs = W3 (h2 + t2g) + b3
    const float* w = W3 + t * 32;
    float s = b3[t];
    for (int k = 0; k < 32; ++k) s = fmaf(w[k], h2[k] + t2g[k], s);
    z[t] = s;
    zs_out[(size_t)row * LDIM + t] = s;
  }
  __syncthreads();

  if (t < 32) {  // decoder layer 1
    const float* w = D1 + t * 32;
    float s = d1[t];
    for (int k = 0; k < 32; ++k) s = fmaf(w[k], z[k], s);
    g1[t] = s > 0.0f ? s : 0.0f;
  }
  __syncthreads();

  {  // decoder layer 2 (64 outputs)
    const float* w = D2 + t * 32;
    float s = d2[t];
    for (int k = 0; k < 32; ++k) s = fmaf(w[k], g1[k], s);
    g2[t] = s > 0.0f ? s : 0.0f;
  }
  __syncthreads();

  // decoder layer 3: 512 outputs, 8 per thread
  for (int q = 0; q < 8; ++q) {
    const int i = t + 64 * q;
    const float* w = D3 + (size_t)i * 64;
    float s = d3[i];
    for (int k = 0; k < 64; ++k) s = fmaf(w[k], g2[k], s);
    xhat[(size_t)row * DIM + i] = s;
  }
}

// ---------------------------------------------------------------------------
extern "C" void kernel_launch(void* const* d_in, const int* in_sizes, int n_in,
                              void* d_out, int out_size, void* d_ws, size_t ws_size,
                              hipStream_t stream) {
  (void)in_sizes; (void)n_in; (void)out_size; (void)ws_size;
  const float* xs = (const float*)d_in[0];
  const float* W1 = (const float*)d_in[1];
  const float* b1 = (const float*)d_in[2];
  const float* W2 = (const float*)d_in[3];
  const float* b2 = (const float*)d_in[4];
  const float* W3 = (const float*)d_in[5];
  const float* b3 = (const float*)d_in[6];
  const float* D1 = (const float*)d_in[7];
  const float* d1 = (const float*)d_in[8];
  const float* D2 = (const float*)d_in[9];
  const float* d2 = (const float*)d_in[10];
  const float* D3 = (const float*)d_in[11];
  const float* d3 = (const float*)d_in[12];

  float* xhat   = (float*)d_out;                       // [N, D]
  float* zs_out = (float*)d_out + (size_t)NROWS * DIM; // [N, L]

  // ws: Xh f16 (8MB) | sq (32KB) | closest (32KB) | pd (128KB) | pi (128KB)
  char* p = (char*)d_ws;
  _Float16* Xh = (_Float16*)p;  p += (size_t)NROWS * DIM * sizeof(_Float16);
  float* sq    = (float*)p;     p += (size_t)NROWS * sizeof(float);
  int* closest = (int*)p;       p += (size_t)NROWS * sizeof(int);
  float* pd    = (float*)p;     p += (size_t)NSPLIT * NROWS * sizeof(float);
  int* pi      = (int*)p;

  prep_kernel<<<NROWS, 256, 0, stream>>>(xs, Xh, sq);
  nn_kernel<<<NSPLIT * 64, 256, 0, stream>>>(Xh, sq, pd, pi);
  nn_reduce_kernel<<<NROWS / 256, 256, 0, stream>>>(pd, pi, closest);
  mlp_kernel<<<NROWS, 64, 0, stream>>>(xs, closest, W1, b1, W2, b2, W3, b3,
                                       D1, d1, D2, d2, D3, d3, xhat, zs_out);
}
